// FinetuneBertFANAttention_73254962201223
// MI455X (gfx1250) — matrix operand order repaired and checked by reading, not compiled, not run on hardware
//
#include <hip/hip_runtime.h>

// ---------------------------------------------------------------------------
// FinetuneBertFANAttention for MI455X (gfx1250, wave32, WMMA)
// B=32, S=512, D=768.  Pipeline:
//   1) convert emb -> bf16, transpose-convert W1,W2 -> bf16 (n-major)
//   2) GEMM1: H1 = relu(X @ W1 + b1)            (bf16 WMMA, bf16 out)
//   3) GEMM2: Xf = emb + (H1 @ W2 + b2)         (bf16 WMMA, f32 out)
//   4) LayerNorm + row norms + out1/out2 heads  (f32 VALU, ff in-place)
//   5) u[b] = sum_j ff[b,j]/||ff[b,j]||         (algebraic collapse of S^2)
//   6) aw[b,j] = <ff[b,j],u[b]>/(||ff||*S); softmax(sigmoid(-aw))
// GEMM: 256x64 block tile, BK=32, double-buffered LDS fed by
// GLOBAL_LOAD_ASYNC_TO_LDS_B128 (ASYNCcnt) when the builtin exists.
// ---------------------------------------------------------------------------

typedef __bf16 bf16_t;
typedef bf16_t v16bf __attribute__((ext_vector_type(16)));
typedef float  v8f   __attribute__((ext_vector_type(8)));
typedef int    i4    __attribute__((ext_vector_type(4)));

#define D_DIM   768
#define S_LEN   512
#define B_BATCH 32
#define NTOK    (B_BATCH * S_LEN)   // 16384

__device__ __forceinline__ unsigned short f2bf(float f) {
  unsigned int u = __float_as_uint(f);
  unsigned int r = (u + 0x7FFFu + ((u >> 16) & 1u)) >> 16;  // RNE
  return (unsigned short)r;
}

// ----------------- 16B global->LDS copy (async if available) ---------------

#if defined(__has_builtin)
#if __has_builtin(__builtin_amdgcn_global_load_async_to_lds_b128)
#define USE_ASYNC_LDS 1
#endif
#endif
#ifndef USE_ASYNC_LDS
#define USE_ASYNC_LDS 0
#endif

#if USE_ASYNC_LDS
// Builtin signature (from hipcc diagnostic): param0 = int4 in AS(1) ("__device__"),
// param1 = int4 in AS(3) ("__shared__"), then imm offset, imm cpol.
typedef __attribute__((address_space(1))) i4 gi4_t;
typedef __attribute__((address_space(3))) i4 li4_t;
template <int IMM>
__device__ __forceinline__ void copy16(const unsigned short* g, unsigned short* l) {
  __builtin_amdgcn_global_load_async_to_lds_b128((gi4_t*)g, (li4_t*)l, IMM, 0);
}
__device__ __forceinline__ void wait_copies() {
#if __has_builtin(__builtin_amdgcn_s_wait_asynccnt)
  __builtin_amdgcn_s_wait_asynccnt(0);
#else
  asm volatile("s_wait_asynccnt 0x0" ::: "memory");
#endif
}
#else
template <int IMM>
__device__ __forceinline__ void copy16(const unsigned short* g, unsigned short* l) {
  *(i4*)((char*)l + IMM) = *(const i4*)((const char*)g + IMM);
}
__device__ __forceinline__ void wait_copies() {}
#endif

// ------------------------- conversion kernels ------------------------------

__global__ __launch_bounds__(256) void cvt_bf16_kernel(
    const float* __restrict__ src, unsigned short* __restrict__ dst, int n) {
  int i = blockIdx.x * 256 + threadIdx.x;
  if (i < n) dst[i] = f2bf(src[i]);
}

// dst[n*768+k] = bf16(src[k*768+n])  (768x768 weight -> n-major)
__global__ __launch_bounds__(256) void transpose_bf16_kernel(
    const float* __restrict__ src, unsigned short* __restrict__ dst) {
  int i = blockIdx.x * 256 + threadIdx.x;
  if (i >= D_DIM * D_DIM) return;
  int n = i / D_DIM, k = i - n * D_DIM;
  dst[i] = f2bf(src[k * D_DIM + n]);
}

// ------------------------------ WMMA GEMM ----------------------------------
// C[M=16384, N=768] = A[M,768] @ W[768,N] (+bias, +relu / +residual)
// Block tile 256x64, K-slice 32, 8 waves: wave w owns rows [32w, 32w+32)
// (2 M-tiles), 4 WMMA 16x16 tiles along N -> 8 WMMA per wave per K-step.
// Double-buffered LDS; next tile staged while current is consumed.
// MODE 0: relu -> bf16 out.  MODE 1: +resid -> f32 out.

template <int MODE>
__global__ __launch_bounds__(256) void gemm_wmma_kernel(
    const unsigned short* __restrict__ A,    // [M,768] bf16 row-major
    const unsigned short* __restrict__ Wt,   // [768,768] bf16, n-major (Wt[n][k])
    const float* __restrict__ bias,          // [768]
    const float* __restrict__ resid,         // [M,768] f32 (MODE 1)
    unsigned short* __restrict__ outBf,      // MODE 0
    float* __restrict__ outF)                // MODE 1
{
  constexpr int BM = 256, BN = 64, BK = 32, LDA = 40, LDB = 40;
  constexpr int NK = D_DIM / BK;             // 24
  __shared__ __align__(16) unsigned short sA[2][BM * LDA];  // 2 x 20 KB
  __shared__ __align__(16) unsigned short sB[2][BN * LDB];  // 2 x  5 KB

  const int tid  = threadIdx.x;
  const int wave = tid >> 5;
  const int lane = tid & 31;
  const int lrow = lane & 15;
  const int m0 = blockIdx.x * BM;
  const int n0 = blockIdx.y * BN;

  // Per-thread copy assignments: A row = tid (64B), B row = tid>>2 seg = tid&3.
  const unsigned short* gA = A + (size_t)(m0 + tid) * D_DIM;
  const int brow = tid >> 2, bseg = tid & 3;
  const unsigned short* gB = Wt + (size_t)(n0 + brow) * D_DIM + bseg * 8;

  auto stage = [&](int buf, int k0) {
    const unsigned short* ga = gA + k0;
    unsigned short* la = &sA[buf][tid * LDA];
    copy16<0>(ga, la);
    copy16<16>(ga, la);
    copy16<32>(ga, la);
    copy16<48>(ga, la);
    copy16<0>(gB + k0, &sB[buf][brow * LDB + bseg * 8]);
  };

  v8f acc[2][4] = {};

  stage(0, 0);
  wait_copies();
  __syncthreads();

  for (int kk = 0; kk < NK; ++kk) {
    const int p = kk & 1;
    if (kk + 1 < NK) stage(1 - p, (kk + 1) * BK);   // prefetch into other buffer

    union { i4 q[2]; v16bf v; } fa[2], fb;
    // A fragment (ISA 16-bit A 16x32 layout):
    //   lanes 0-15: K {0..7} then {16..23}; lanes 16-31: K {8..15} then {24..31}
    const int aoff = (lane < 16) ? 0 : 8;
#pragma unroll
    for (int mt = 0; mt < 2; ++mt) {
      const unsigned short* ap = &sA[p][(wave * 32 + mt * 16 + lrow) * LDA + aoff];
      fa[mt].q[0] = *(const i4*)(ap);
      fa[mt].q[1] = *(const i4*)(ap + 16);
    }
    // B fragment: lane n holds col n;
    //   lanes 0-15: K {0..7} then {16..23}; lanes 16-31: K {8..15} then {24..31}
    const int boff = (lane < 16) ? 0 : 8;
#pragma unroll
    for (int nb = 0; nb < 4; ++nb) {
      const unsigned short* bp = &sB[p][(nb * 16 + lrow) * LDB + boff];
      fb.q[0] = *(const i4*)(bp);
      fb.q[1] = *(const i4*)(bp + 16);
#pragma unroll
      for (int mt = 0; mt < 2; ++mt)
        acc[mt][nb] = __builtin_amdgcn_wmma_f32_16x16x32_bf16(
            false, fa[mt].v, false, fb.v, (short)0, acc[mt][nb], false, false);
    }
    wait_copies();      // next buffer filled
    __syncthreads();    // everyone done reading current buffer
  }

  // Epilogue. C layout: VGPR r -> (M = r + (lane>=16 ? 8:0), N = lane&15)
#pragma unroll
  for (int mt = 0; mt < 2; ++mt) {
#pragma unroll
    for (int nb = 0; nb < 4; ++nb) {
      int col = n0 + nb * 16 + lrow;
      float bv = bias[col];
#pragma unroll
      for (int r = 0; r < 8; ++r) {
        int m = m0 + wave * 32 + mt * 16 + r + ((lane >= 16) ? 8 : 0);
        size_t idx = (size_t)m * D_DIM + col;
        float v = acc[mt][nb][r] + bv;
        if (MODE == 0) {
          outBf[idx] = f2bf(fmaxf(v, 0.0f));
        } else {
          outF[idx] = v + resid[idx];
        }
      }
    }
  }
}

// ------------------------- LayerNorm + heads -------------------------------

__device__ __forceinline__ float blockReduce256(float v, float* red) {
  int t = threadIdx.x;
  red[t] = v;
  __syncthreads();
#pragma unroll
  for (int s = 128; s > 0; s >>= 1) {
    if (t < s) red[t] += red[t + s];
    __syncthreads();
  }
  float r = red[0];
  __syncthreads();
  return r;
}

__global__ __launch_bounds__(256) void ln_heads_kernel(
    float* __restrict__ x,                    // in: emb+h2, out: ff (in-place)
    const float* __restrict__ gamma, const float* __restrict__ beta,
    const float* __restrict__ Wy, const float* __restrict__ by,
    const float* __restrict__ Wz, const float* __restrict__ bz,
    float* __restrict__ norms, float* __restrict__ out)
{
  __shared__ float red[256];
  const int row = blockIdx.x, t = threadIdx.x;
  float* xr = x + (size_t)row * D_DIM;

  int   d[3];
  float v[3];
#pragma unroll
  for (int i = 0; i < 3; ++i) { d[i] = t + i * 256; v[i] = xr[d[i]]; }

  float mu = blockReduce256(v[0] + v[1] + v[2], red) * (1.0f / D_DIM);
  float sq = 0.f;
#pragma unroll
  for (int i = 0; i < 3; ++i) { float c = v[i] - mu; sq += c * c; }
  float var  = blockReduce256(sq, red) * (1.0f / D_DIM);
  float rstd = rsqrtf(var + 1e-5f);

  float f[3]; float nsq = 0.f;
#pragma unroll
  for (int i = 0; i < 3; ++i) {
    f[i] = (v[i] - mu) * rstd * gamma[d[i]] + beta[d[i]];
    xr[d[i]] = f[i];
    nsq += f[i] * f[i];
  }
  float tot = blockReduce256(nsq, red);
  if (t == 0) norms[row] = sqrtf(tot);

  // skinny heads: out1 = ff@Wy + by (3), out2 = ff@Wz + bz (5)
  float p[8];
#pragma unroll
  for (int o = 0; o < 8; ++o) p[o] = 0.f;
#pragma unroll
  for (int i = 0; i < 3; ++i) {
#pragma unroll
    for (int y = 0; y < 3; ++y) p[y]     += f[i] * Wy[d[i] * 3 + y];
#pragma unroll
    for (int z = 0; z < 5; ++z) p[3 + z] += f[i] * Wz[d[i] * 5 + z];
  }
  float rr[8];
  for (int o = 0; o < 8; ++o) rr[o] = blockReduce256(p[o], red);
  if (t == 0) {
    for (int y = 0; y < 3; ++y) out[(size_t)row * 3 + y] = rr[y] + by[y];
    for (int z = 0; z < 5; ++z)
      out[(size_t)NTOK * 3 + (size_t)row * 5 + z] = rr[3 + z] + bz[z];
  }
}

// ---------------- cosine-attention via u-vector collapse -------------------

__global__ __launch_bounds__(256) void u_kernel(
    const float* __restrict__ ff, const float* __restrict__ norms,
    float* __restrict__ u) {
  int b = blockIdx.x, t = threadIdx.x;
  float a0 = 0.f, a1 = 0.f, a2 = 0.f;
  for (int j = 0; j < S_LEN; ++j) {
    float inv = 1.0f / norms[b * S_LEN + j];
    const float* r = ff + (size_t)(b * S_LEN + j) * D_DIM;
    a0 += r[t] * inv;
    a1 += r[t + 256] * inv;
    a2 += r[t + 512] * inv;
  }
  u[(size_t)b * D_DIM + t]       = a0;
  u[(size_t)b * D_DIM + t + 256] = a1;
  u[(size_t)b * D_DIM + t + 512] = a2;
}

__global__ __launch_bounds__(512) void attn_kernel(
    const float* __restrict__ ff, const float* __restrict__ norms,
    const float* __restrict__ u, float* __restrict__ out) {
  __shared__ float red[512];
  int b = blockIdx.x, j = threadIdx.x;
  const float* r  = ff + (size_t)(b * S_LEN + j) * D_DIM;
  const float* ub = u + (size_t)b * D_DIM;
  float dot = 0.f;
  for (int dd = 0; dd < D_DIM; ++dd) dot += r[dd] * ub[dd];
  float aw = dot / (norms[b * S_LEN + j] * (float)S_LEN);
  float sg = 1.0f / (1.0f + __expf(aw));   // sigmoid(-aw)

  // softmax over S
  red[j] = sg; __syncthreads();
  for (int s = 256; s > 0; s >>= 1) {
    if (j < s) red[j] = fmaxf(red[j], red[j + s]);
    __syncthreads();
  }
  float mx = red[0]; __syncthreads();
  float e = __expf(sg - mx);
  red[j] = e; __syncthreads();
  for (int s = 256; s > 0; s >>= 1) {
    if (j < s) red[j] += red[j + s];
    __syncthreads();
  }
  float sum = red[0];
  out[(size_t)NTOK * 8 + (size_t)b * S_LEN + j] = e / sum;
}

// ------------------------------- launch ------------------------------------

extern "C" void kernel_launch(void* const* d_in, const int* in_sizes, int n_in,
                              void* d_out, int out_size, void* d_ws, size_t ws_size,
                              hipStream_t stream) {
  (void)in_sizes; (void)n_in; (void)out_size; (void)ws_size;

  const float* emb   = (const float*)d_in[0];
  const float* W1    = (const float*)d_in[1];
  const float* b1    = (const float*)d_in[2];
  const float* W2    = (const float*)d_in[3];
  const float* b2    = (const float*)d_in[4];
  const float* gamma = (const float*)d_in[5];
  const float* beta  = (const float*)d_in[6];
  const float* Wy    = (const float*)d_in[7];
  const float* by    = (const float*)d_in[8];
  const float* Wz    = (const float*)d_in[9];
  const float* bz    = (const float*)d_in[10];
  float* out = (float*)d_out;

  // workspace carve
  char* p = (char*)d_ws;
  unsigned short* Xbf  = (unsigned short*)p; p += (size_t)NTOK * D_DIM * 2;   // 25.2 MB
  unsigned short* W1t  = (unsigned short*)p; p += (size_t)D_DIM * D_DIM * 2;  // 1.2 MB
  unsigned short* W2t  = (unsigned short*)p; p += (size_t)D_DIM * D_DIM * 2;  // 1.2 MB
  unsigned short* H1bf = (unsigned short*)p; p += (size_t)NTOK * D_DIM * 2;   // 25.2 MB
  float*          Xf   = (float*)p;          p += (size_t)NTOK * D_DIM * 4;   // 50.3 MB
  float*          norms= (float*)p;          p += (size_t)NTOK * 4;
  float*          u    = (float*)p;          p += (size_t)B_BATCH * D_DIM * 4;

  const int nElem = NTOK * D_DIM;                 // 12,582,912
  cvt_bf16_kernel<<<(nElem + 255) / 256, 256, 0, stream>>>(emb, Xbf, nElem);
  const int nW = D_DIM * D_DIM;                   // 589,824
  transpose_bf16_kernel<<<(nW + 255) / 256, 256, 0, stream>>>(W1, W1t);
  transpose_bf16_kernel<<<(nW + 255) / 256, 256, 0, stream>>>(W2, W2t);

  dim3 ggrid(NTOK / 256, D_DIM / 64);             // 64 x 12
  gemm_wmma_kernel<0><<<ggrid, 256, 0, stream>>>(Xbf,  W1t, b1, nullptr, H1bf, nullptr);
  gemm_wmma_kernel<1><<<ggrid, 256, 0, stream>>>(H1bf, W2t, b2, emb,     nullptr, Xf);

  ln_heads_kernel<<<NTOK, 256, 0, stream>>>(Xf, gamma, beta, Wy, by, Wz, bz, norms, out);
  u_kernel<<<B_BATCH, 256, 0, stream>>>(Xf, norms, u);
  attn_kernel<<<B_BATCH, 512, 0, stream>>>(Xf, norms, u, out);
}
